// Model_40295383171604
// MI455X (gfx1250) — matrix operand order repaired and checked by reading, not compiled, not run on hardware
//
#include <hip/hip_runtime.h>

#define VOCAB  4711
#define HID    1024
#define G4     4096      // 4*HID
#define BATCH  64
#define SEQ    150
#define NENC   50
#define NDEC   101
#define VPAD   4736      // vocab padded to multiple of 128
#define EOS    4710

typedef __bf16 v16bf __attribute__((ext_vector_type(16)));
typedef float  v8f   __attribute__((ext_vector_type(8)));

union BFrag { v16bf v; uint4 q[2]; };

__device__ __forceinline__ unsigned short f2bf(float f) {
  union { float f; unsigned u; } x; x.f = f;
  unsigned u = x.u;
  u += 0x7FFFu + ((u >> 16) & 1u);     // round to nearest even
  return (unsigned short)(u >> 16);
}

// ---------------- prep kernels ----------------

// W [N][K] fp32 row-major  ->  Wt [K][NP] bf16 row-major (transpose + convert, zero-pad N..NP)
__global__ void k_transpose(const float* __restrict__ W, unsigned short* __restrict__ Wt,
                            int N, int K, int NP) {
  __shared__ float tile[32][33];
  const int kb = blockIdx.x * 32, nb = blockIdx.y * 32;
  const int tx = threadIdx.x & 31, ty = threadIdx.x >> 5;   // 32 x 8
#pragma unroll
  for (int i = 0; i < 4; ++i) {
    int n = nb + ty + 8 * i, k = kb + tx;
    float v = 0.f;
    if (n < N && k < K) v = W[(size_t)n * K + k];
    tile[ty + 8 * i][tx] = v;
  }
  __syncthreads();
#pragma unroll
  for (int i = 0; i < 4; ++i) {
    int k = kb + ty + 8 * i, n = nb + tx;
    if (k < K && n < NP) Wt[(size_t)k * NP + n] = f2bf(tile[tx][ty + 8 * i]);
  }
}

__global__ void k_init(float* c, unsigned short* hb0) {
  int i = blockIdx.x * blockDim.x + threadIdx.x;     // 65536
  c[i] = 0.f; hb0[i] = 0;
}
__global__ void k_zero_c(float* c) {
  int i = blockIdx.x * blockDim.x + threadIdx.x;
  c[i] = 0.f;
}

// Pre-gather x @ W_ih.T for every (step, batch): xw[s][b][j] = W_ih[j][tok(s,b)]
__global__ void k_gather(const float* __restrict__ W_ih, const int* __restrict__ tokens,
                         float* __restrict__ xw, int is_dec) {
  const int s = blockIdx.x >> 6;
  const int b = blockIdx.x & 63;
  const int col = is_dec ? (NENC - 1 + s) : (NENC - 1 - s);
  const int tok = tokens[b * SEQ + col];
  float* dst = xw + ((size_t)s * BATCH + b) * (size_t)G4;
  for (int j = threadIdx.x; j < G4; j += blockDim.x)
    dst[j] = W_ih[(size_t)j * VOCAB + tok];
}

// ---------------- fused LSTM step: gates GEMM (split-K WMMA) + cell, h ping-pong -------------
// Block: 256 threads = 8 waves = 2 j-tiles x 2 M-halves x 2 K-halves. Grid: 32 blocks.
// Each wave: 8 accumulators (2 batch M-tiles x 4 gate groups, 64 VGPRs) over K/2 = 128 WMMAs.
// ks=1 wave spills partials to LDS; ks=0 wave reduces and applies the LSTM cell for its
// 32 batch rows x 16 j columns (all four gates are wave-local by construction).
__global__ void __launch_bounds__(256, 1)
k_step(const unsigned short* __restrict__ hbin,  // h_prev bf16 [64][1024]
       const unsigned short* __restrict__ Wt,    // [1024][4096] bf16
       const float* __restrict__ xw,             // [64][4096] this step, or nullptr
       const float* __restrict__ W_ih,           // direct-gather fallback
       const int* __restrict__ tokens, int tok_col,
       const float* __restrict__ b_ih, const float* __restrict__ b_hh,
       float* __restrict__ c,
       unsigned short* __restrict__ hbout) {
  __shared__ float red[2][2][8][8][32];                // 32 KB
  const int wave = threadIdx.x >> 5;
  const int lane = threadIdx.x & 31;
  const int jl   = wave >> 2;                          // local j-tile 0..1
  const int mh   = (wave >> 1) & 1;                    // M half 0..1
  const int ks   = wave & 1;                           // K split
  const int jt   = blockIdx.x * 32 + jl * 16;
  const int row  = lane & 15;
  const int hiK  = lane >> 4;

  v8f acc[8];                                          // [m_local*4 + g]
#pragma unroll
  for (int a = 0; a < 8; ++a)
#pragma unroll
    for (int r = 0; r < 8; ++r) acc[a][r] = 0.f;

  const int kbeg = ks * (HID / 2);
  for (int k0 = kbeg; k0 < kbeg + HID / 2; k0 += 32) {
    BFrag bf[4];
#pragma unroll
    for (int g = 0; g < 4; ++g) {                       // B: Wt col g*1024+jt+row, rows k0+8*hiK+s (+8 for s>=8)
      const unsigned short* bp = Wt + (size_t)(k0 + (hiK << 3)) * G4 + g * HID + jt + row;
#pragma unroll
      for (int s = 0; s < 16; ++s)
        ((unsigned short*)&bf[g])[s] = bp[(size_t)(s + (s >= 8 ? 8 : 0)) * G4];
    }
#pragma unroll
    for (int ml = 0; ml < 2; ++ml) {                    // A: h rows (mh*2+ml)*16 + row
      BFrag af;
      const unsigned short* ap =
          hbin + (size_t)(((mh * 2 + ml) << 4) + row) * HID + k0 + (hiK << 3);
      af.q[0] = *(const uint4*)(ap);
      af.q[1] = *(const uint4*)(ap + 16);
#pragma unroll
      for (int g = 0; g < 4; ++g)
        acc[ml * 4 + g] = __builtin_amdgcn_wmma_f32_16x16x32_bf16(
            false, af.v, false, bf[g].v, (short)0, acc[ml * 4 + g], false, false);
    }
  }

  if (ks == 1) {
#pragma unroll
    for (int a = 0; a < 8; ++a)
#pragma unroll
      for (int r = 0; r < 8; ++r) red[jl][mh][a][r][lane] = acc[a][r];
  }
  __syncthreads();
  if (ks == 1) return;                                 // no barriers below

  // ---- reduce split-K partials, then LSTM cell on this wave's 32 x 16 strip ----
  const int j = jt + row;                              // column owned by this lane
  float bi[4], bh[4];
#pragma unroll
  for (int g = 0; g < 4; ++g) { bi[g] = b_ih[g * HID + j]; bh[g] = b_hh[g * HID + j]; }

#pragma unroll
  for (int ml = 0; ml < 2; ++ml) {
#pragma unroll
    for (int r = 0; r < 8; ++r) {
      const int b = ((mh * 2 + ml) << 4) + (hiK << 3) + r;  // batch row (C-fragment layout)
      float gi = acc[ml * 4 + 0][r] + red[jl][mh][ml * 4 + 0][r][lane];
      float gf = acc[ml * 4 + 1][r] + red[jl][mh][ml * 4 + 1][r][lane];
      float gg = acc[ml * 4 + 2][r] + red[jl][mh][ml * 4 + 2][r][lane];
      float go = acc[ml * 4 + 3][r] + red[jl][mh][ml * 4 + 3][r][lane];
      float xi, xf, xg, xo;
      if (xw) {
        const float* x = xw + (size_t)b * G4 + j;
        xi = x[0]; xf = x[HID]; xg = x[2 * HID]; xo = x[3 * HID];
      } else {
        const int tok = tokens[b * SEQ + tok_col];
        xi = W_ih[(size_t)(j)           * VOCAB + tok];
        xf = W_ih[(size_t)(j + HID)     * VOCAB + tok];
        xg = W_ih[(size_t)(j + 2 * HID) * VOCAB + tok];
        xo = W_ih[(size_t)(j + 3 * HID) * VOCAB + tok];
      }
      gi += xi + bi[0] + bh[0];
      gf += xf + bi[1] + bh[1];
      gg += xg + bi[2] + bh[2];
      go += xo + bi[3] + bh[3];
      const float si = 1.f / (1.f + expf(-gi));
      const float sf = 1.f / (1.f + expf(-gf));
      const float so = 1.f / (1.f + expf(-go));
      const float cn = sf * c[b * HID + j] + si * tanhf(gg);
      const float hn = so * tanhf(cn);
      c[b * HID + j] = cn;
      hbout[b * HID + j] = f2bf(hn);
    }
  }
}

// ---------------- WMMA GEMM for logits: out[64][NN] = hb @ Wt[1024][NN] ----------------
__global__ void __launch_bounds__(256, 1)
k_gemm(const unsigned short* __restrict__ hb,
       const unsigned short* __restrict__ Wt,
       float* __restrict__ out, int NN) {
  const int wave = threadIdx.x >> 5;
  const int lane = threadIdx.x & 31;
  const int n0 = (blockIdx.x * 8 + wave) << 4;
  if (n0 >= NN) return;                      // wave-uniform
  const int row = lane & 15;
  const int hiK = lane >> 4;

  v8f acc[4];
#pragma unroll
  for (int m = 0; m < 4; ++m)
#pragma unroll
    for (int r = 0; r < 8; ++r) acc[m][r] = 0.f;

  for (int k0 = 0; k0 < HID; k0 += 32) {
    BFrag bf;
    const unsigned short* bp = Wt + (size_t)(k0 + (hiK << 3)) * NN + n0 + row;
#pragma unroll
    for (int s = 0; s < 16; ++s)
      ((unsigned short*)&bf)[s] = bp[(size_t)(s + (s >= 8 ? 8 : 0)) * NN];
#pragma unroll
    for (int m = 0; m < 4; ++m) {
      BFrag af;
      const unsigned short* ap = hb + (size_t)((m << 4) + row) * HID + k0 + (hiK << 3);
      af.q[0] = *(const uint4*)(ap);
      af.q[1] = *(const uint4*)(ap + 16);
      acc[m] = __builtin_amdgcn_wmma_f32_16x16x32_bf16(
          false, af.v, false, bf.v, (short)0, acc[m], false, false);
    }
  }
  const int nc = n0 + row;
  const int mb = hiK << 3;
#pragma unroll
  for (int m = 0; m < 4; ++m)
#pragma unroll
    for (int r = 0; r < 8; ++r)
      out[(size_t)((m << 4) + mb + r) * NN + nc] = acc[m][r];
}

// ---------------- per-step loss: loss[b] = logsumexp(logits) - logits[tgt] ----------------
__global__ void k_loss(const float* __restrict__ Lg,     // [64][VPAD]
                       const float* __restrict__ out_b,
                       const int* __restrict__ tokens, int tgt_col,  // -1 => EOS
                       float* __restrict__ loss) {
  const int b = blockIdx.x, t = threadIdx.x;
  __shared__ float red[256];
  float s = 0.f;
  const float* L = Lg + (size_t)b * VPAD;
  for (int j = t; j < VOCAB; j += 256) s += expf(L[j] + out_b[j]);
  red[t] = s; __syncthreads();
  for (int w = 128; w; w >>= 1) { if (t < w) red[t] += red[t + w]; __syncthreads(); }
  if (t == 0) {
    const int tgt = (tgt_col >= 0) ? tokens[b * SEQ + tgt_col] : EOS;
    loss[b] = logf(red[0]) - (L[tgt] + out_b[tgt]);
  }
}

__global__ void k_final(const float* __restrict__ loss, float* __restrict__ out) {
  __shared__ float red[256];
  float s = 0.f;
  for (int i = threadIdx.x; i < NDEC * BATCH; i += 256) s += loss[i];
  red[threadIdx.x] = s; __syncthreads();
  for (int w = 128; w; w >>= 1) { if (threadIdx.x < w) red[threadIdx.x] += red[threadIdx.x + w]; __syncthreads(); }
  if (threadIdx.x == 0) out[0] = red[0] / (float)BATCH;
}

// ---------------- host ----------------
extern "C" void kernel_launch(void* const* d_in, const int* in_sizes, int n_in,
                              void* d_out, int out_size, void* d_ws, size_t ws_size,
                              hipStream_t stream) {
  const int*   tokens = (const int*)d_in[0];
  const float* eWih = (const float*)d_in[1];
  const float* eWhh = (const float*)d_in[2];
  const float* ebih = (const float*)d_in[3];
  const float* ebhh = (const float*)d_in[4];
  const float* dWih = (const float*)d_in[5];
  const float* dWhh = (const float*)d_in[6];
  const float* dbih = (const float*)d_in[7];
  const float* dbhh = (const float*)d_in[8];
  const float* oW   = (const float*)d_in[9];
  const float* ob   = (const float*)d_in[10];
  float* dout = (float*)d_out;

  char* ws = (char*)d_ws;
  size_t off = 0;
  auto carve = [&](size_t bytes) -> char* {
    char* p = ws + off; off = (off + bytes + 255) & ~(size_t)255; return p;
  };
  unsigned short* WtE = (unsigned short*)carve((size_t)HID * G4 * 2);
  unsigned short* WtD = (unsigned short*)carve((size_t)HID * G4 * 2);
  unsigned short* WtO = (unsigned short*)carve((size_t)HID * VPAD * 2);
  float*          C   = (float*)carve((size_t)BATCH * HID * 4);
  unsigned short* HB0 = (unsigned short*)carve((size_t)BATCH * HID * 2);
  unsigned short* HB1 = (unsigned short*)carve((size_t)BATCH * HID * 2);
  float*          LG  = (float*)carve((size_t)BATCH * VPAD * 4);
  float*          LS  = (float*)carve((size_t)NDEC * BATCH * 4);
  float*          XWE = (float*)carve((size_t)NENC * BATCH * G4 * 4);
  float*          XWD = (float*)carve((size_t)NDEC * BATCH * G4 * 4);
  const bool pregather = (ws_size >= off);   // deterministic launch-time choice
  unsigned short* HB[2] = { HB0, HB1 };

  // --- prep: bf16 transposed weights, zeroed state, hoisted one-hot gathers ---
  k_transpose<<<dim3(HID / 32, G4 / 32),   256, 0, stream>>>(eWhh, WtE, G4,    HID, G4);
  k_transpose<<<dim3(HID / 32, G4 / 32),   256, 0, stream>>>(dWhh, WtD, G4,    HID, G4);
  k_transpose<<<dim3(HID / 32, VPAD / 32), 256, 0, stream>>>(oW,   WtO, VOCAB, HID, VPAD);
  k_init<<<256, 256, 0, stream>>>(C, HB0);
  if (pregather) {
    k_gather<<<NENC * BATCH, 256, 0, stream>>>(eWih, tokens, XWE, 0);
    k_gather<<<NDEC * BATCH, 256, 0, stream>>>(dWih, tokens, XWD, 1);
  }

  // --- encoder: 50 serial fused steps (h ping-pong; h_enc ends in HB[0] since 50 is even) ---
  for (int s = 0; s < NENC; ++s) {
    k_step<<<32, 256, 0, stream>>>(HB[s & 1], WtE,
                                   pregather ? XWE + (size_t)s * BATCH * G4 : nullptr,
                                   eWih, tokens, NENC - 1 - s, ebih, ebhh,
                                   C, HB[(s + 1) & 1]);
  }
  k_zero_c<<<256, 256, 0, stream>>>(C);   // decoder starts with (h_enc, c=0)

  // --- decoder: 101 serial fused steps + logits WMMA GEMM + CE loss ---
  for (int s = 0; s < NDEC; ++s) {
    k_step<<<32, 256, 0, stream>>>(HB[s & 1], WtD,
                                   pregather ? XWD + (size_t)s * BATCH * G4 : nullptr,
                                   dWih, tokens, NENC - 1 + s, dbih, dbhh,
                                   C, HB[(s + 1) & 1]);
    k_gemm<<<VPAD / 128, 256, 0, stream>>>(HB[(s + 1) & 1], WtO, LG, VPAD);
    k_loss<<<BATCH, 256, 0, stream>>>(LG, ob, tokens,
                                      (s < NDEC - 1) ? (NENC + s) : -1,
                                      LS + (size_t)s * BATCH);
  }
  k_final<<<1, 256, 0, stream>>>(LS, dout);
}
